// RelativeMultiHeadAttention_56985626083937
// MI455X (gfx1250) — matrix operand order repaired and checked by reading, not compiled, not run on hardware
//
#include <hip/hip_runtime.h>
#include <hip/hip_bf16.h>

typedef __attribute__((ext_vector_type(16))) _Float16 v16h;
typedef __attribute__((ext_vector_type(8)))  _Float16 v8h;
typedef __attribute__((ext_vector_type(4)))  _Float16 v4h;
typedef __attribute__((ext_vector_type(8)))  float    v8f;

union F16x16 {
    v16h v;
    struct { v8h lo, hi; } p;
};

#define HIDDEN   4096
#define HEADS    32
#define HEAD_DIM 128
#define NB       4
#define SQ       2048
#define SK       128

#define TILE_K     32
#define LDS_STRIDE 40   // 32 + 8 f16 pad (80 bytes/row, keeps 16B alignment)

// ---------------------------------------------------------------------------
// WMMA helper: D = A(16x32 f16) * B(32x16 f16) + C(16x16 f32)
// ---------------------------------------------------------------------------
__device__ __forceinline__ v8f wmma16(v16h a, v16h b, v8f c) {
    return __builtin_amdgcn_wmma_f32_16x16x32_f16(false, a, false, b,
                                                  (short)0, c, false, false);
}

// A-fragment (16x32, MxK): lane l -> row l%16; elems[0..7]=K+{0|8}+0..7,
// elems[8..15]=K+16+{0|8}+0..7   (ISA 7.12.2, 16-bit A 16x32)
__device__ __forceinline__ v16h frag_a(const _Float16* p, int stride, int lane) {
    int r  = lane & 15;
    int kh = (lane >> 4) << 3;   // 0 or 8
    F16x16 u;
    u.p.lo = *(const v8h*)(p + (size_t)r * stride + kh);
    u.p.hi = *(const v8h*)(p + (size_t)r * stride + kh + 16);
    return u.v;
}

// B-fragment (32x16, KxN) from K-contiguous (transposed) storage Bt[n][k]:
// lane l -> col l%16; elems[0..7]=K+{0|8}+0..7, elems[8..15]=K+16+{0|8}+0..7
__device__ __forceinline__ v16h frag_b(const _Float16* p, int stride, int lane) {
    int c  = lane & 15;
    int kh = (lane >> 4) << 3;   // 0 or 8
    F16x16 u;
    u.p.lo = *(const v8h*)(p + (size_t)c * stride + kh);
    u.p.hi = *(const v8h*)(p + (size_t)c * stride + kh + 16);
    return u.v;
}

// ---------------------------------------------------------------------------
// f32 -> f16 elementwise convert (n multiple of 4)
// ---------------------------------------------------------------------------
__global__ __launch_bounds__(256) void cvt_f32_f16(const float* __restrict__ in,
                                                   _Float16* __restrict__ out,
                                                   int n) {
    int i = (blockIdx.x * 256 + threadIdx.x) * 4;
    if (i + 3 < n) {
        float4 f = *(const float4*)(in + i);
        v4h o = { (_Float16)f.x, (_Float16)f.y, (_Float16)f.z, (_Float16)f.w };
        *(v4h*)(out + i) = o;
    }
}

// ---------------------------------------------------------------------------
// W[K][N] f32 -> Wt[N][K] f16 (tiled transpose; dims multiples of 32)
// ---------------------------------------------------------------------------
__global__ __launch_bounds__(256) void cvt_transpose(const float* __restrict__ W,
                                                     _Float16* __restrict__ Wt,
                                                     int rows, int cols) {
    __shared__ float t[32][33];
    int x  = blockIdx.x * 32 + threadIdx.x;   // col in W
    int y0 = blockIdx.y * 32;                 // row base in W
#pragma unroll
    for (int i = 0; i < 32; i += 8)
        t[threadIdx.y + i][threadIdx.x] = W[(size_t)(y0 + threadIdx.y + i) * cols + x];
    __syncthreads();
    int xo = blockIdx.y * 32 + threadIdx.x;   // col in Wt (= row of W)
    int yo = blockIdx.x * 32;                 // row base in Wt (= col of W)
#pragma unroll
    for (int i = 0; i < 32; i += 8)
        Wt[(size_t)(yo + threadIdx.y + i) * rows + xo] =
            (_Float16)t[threadIdx.x][threadIdx.y + i];
}

// ---------------------------------------------------------------------------
// C[M][N] = A[M][K](f16) * Bt[N][K](f16)^T
// Block tile 128x256, 256 threads = 8 waves in 2x4 grid, wave tile 64x64
// (4x4 WMMA accums -> 16 wmma : 16 ds_load_b128 per K-step).
// Double-buffered LDS, one barrier per K-step; global loads for step n+1
// issued before the WMMA burst of step n. M mult of 128, N mult of 256,
// K mult of 32.
// ---------------------------------------------------------------------------
template <typename OutT>
__global__ __launch_bounds__(256) void gemm_f16_t(const _Float16* __restrict__ A,
                                                  const _Float16* __restrict__ Bt,
                                                  OutT* __restrict__ C,
                                                  int M, int N, int K) {
    __shared__ _Float16 as[2][128 * LDS_STRIDE];
    __shared__ _Float16 bs[2][256 * LDS_STRIDE];

    const int tid  = threadIdx.x;
    const int lane = tid & 31;
    const int wave = tid >> 5;
    const int wm   = wave >> 2;   // 0..1 -> 64-row band
    const int wn   = wave & 3;    // 0..3 -> 64-col band

    const int tiles_n = N >> 8;
    const int bm = blockIdx.x / tiles_n;
    const int bn = blockIdx.x % tiles_n;
    const int row0 = bm * 128;
    const int col0 = bn * 256;

    v8f acc[4][4];
#pragma unroll
    for (int i = 0; i < 4; ++i)
#pragma unroll
        for (int j = 0; j < 4; ++j) acc[i][j] = v8f{};

    const int lrow = tid >> 2;        // 0..63
    const int lc8  = (tid & 3) << 3;  // 0,8,16,24

    const _Float16* gA = A  + (size_t)(row0 + lrow) * K + lc8;
    const _Float16* gB = Bt + (size_t)(col0 + lrow) * K + lc8;
    const size_t rs = (size_t)64 * K;   // 64-row stride

    // ---- prologue: stage K-step 0 into buffer 0 ----
    v8h ra0 = *(const v8h*)(gA);
    v8h ra1 = *(const v8h*)(gA + rs);
    v8h rb0 = *(const v8h*)(gB);
    v8h rb1 = *(const v8h*)(gB + rs);
    v8h rb2 = *(const v8h*)(gB + 2 * rs);
    v8h rb3 = *(const v8h*)(gB + 3 * rs);
    *(v8h*)(&as[0][(size_t)lrow * LDS_STRIDE + lc8])         = ra0;
    *(v8h*)(&as[0][(size_t)(lrow + 64) * LDS_STRIDE + lc8])  = ra1;
    *(v8h*)(&bs[0][(size_t)lrow * LDS_STRIDE + lc8])         = rb0;
    *(v8h*)(&bs[0][(size_t)(lrow + 64) * LDS_STRIDE + lc8])  = rb1;
    *(v8h*)(&bs[0][(size_t)(lrow + 128) * LDS_STRIDE + lc8]) = rb2;
    *(v8h*)(&bs[0][(size_t)(lrow + 192) * LDS_STRIDE + lc8]) = rb3;
    __syncthreads();

    for (int kb = 0; kb < K; kb += TILE_K) {
        const int  cur  = (kb >> 5) & 1;
        const bool more = (kb + TILE_K) < K;

        // issue next tile's global loads before the WMMA burst
        if (more) {
            const _Float16* pa = gA + kb + TILE_K;
            const _Float16* pb = gB + kb + TILE_K;
            ra0 = *(const v8h*)pa;
            ra1 = *(const v8h*)(pa + rs);
            rb0 = *(const v8h*)pb;
            rb1 = *(const v8h*)(pb + rs);
            rb2 = *(const v8h*)(pb + 2 * rs);
            rb3 = *(const v8h*)(pb + 3 * rs);
            if (kb + 2 * TILE_K < K) {      // gfx1250: global_prefetch_b8
                __builtin_prefetch(pa + TILE_K, 0, 1);
                __builtin_prefetch(pb + TILE_K, 0, 1);
            }
        }

        // compute 64x64 from current buffer: A-frags reused over 4 B columns
        v16h af[4];
#pragma unroll
        for (int i = 0; i < 4; ++i)
            af[i] = frag_a(&as[cur][(size_t)(wm * 64 + i * 16) * LDS_STRIDE],
                           LDS_STRIDE, lane);
#pragma unroll
        for (int j = 0; j < 4; ++j) {
            v16h bf = frag_b(&bs[cur][(size_t)(wn * 64 + j * 16) * LDS_STRIDE],
                             LDS_STRIDE, lane);
#pragma unroll
            for (int i = 0; i < 4; ++i)
                acc[i][j] = wmma16(af[i], bf, acc[i][j]);
        }

        // stage next tile into the other buffer
        if (more) {
            const int nxt = cur ^ 1;
            *(v8h*)(&as[nxt][(size_t)lrow * LDS_STRIDE + lc8])         = ra0;
            *(v8h*)(&as[nxt][(size_t)(lrow + 64) * LDS_STRIDE + lc8])  = ra1;
            *(v8h*)(&bs[nxt][(size_t)lrow * LDS_STRIDE + lc8])         = rb0;
            *(v8h*)(&bs[nxt][(size_t)(lrow + 64) * LDS_STRIDE + lc8])  = rb1;
            *(v8h*)(&bs[nxt][(size_t)(lrow + 128) * LDS_STRIDE + lc8]) = rb2;
            *(v8h*)(&bs[nxt][(size_t)(lrow + 192) * LDS_STRIDE + lc8]) = rb3;
        }
        __syncthreads();
    }

    // C/D layout: vgpr e -> row e + (lane>=16 ? 8 : 0); col = lane&15
    const int radd = (lane >> 4) << 3;
    const int cn   = lane & 15;
#pragma unroll
    for (int i = 0; i < 4; ++i)
#pragma unroll
        for (int j = 0; j < 4; ++j) {
            int rb = row0 + wm * 64 + i * 16 + radd;
            int cc = col0 + wn * 64 + j * 16 + cn;
#pragma unroll
            for (int e = 0; e < 8; ++e)
                C[(size_t)(rb + e) * N + cc] = (OutT)acc[i][j][e];
        }
}

// ---------------------------------------------------------------------------
// Fused attention per (b, h, 128-query-row tile).
// scores = qh*khT*scale + rh ; softmax ; ctx = P*vh  (all SK=d=128)
// ---------------------------------------------------------------------------
__global__ __launch_bounds__(256) void attn_kernel(const _Float16* __restrict__ qh,
                                                   const _Float16* __restrict__ kh,
                                                   const _Float16* __restrict__ vh,
                                                   const _Float16* __restrict__ rh,
                                                   _Float16* __restrict__ ctx) {
    __shared__ _Float16 kv[128 * 128];      // kh[key][d], later vh^T[d->col][key]
    __shared__ _Float16 ps[8][16 * 128];    // per-wave probabilities

    const int tid  = threadIdx.x;
    const int lane = tid & 31;
    const int wave = tid >> 5;

    int bid = blockIdx.x;
    const int qt = bid & 15;  bid >>= 4;    // 16 q-tiles of 128 rows
    const int h  = bid & 31;  bid >>= 5;
    const int b  = bid;                     // 0..3
    const int hcol = h * HEAD_DIM;
    const size_t ld = HIDDEN;

    // stage kh tile [128 keys][128 d] for this (b,h)
    for (int it = 0; it < 8; ++it) {
        int id = tid + it * 256;
        int rr = id >> 4, cc = (id & 15) << 3;
        *(v8h*)(kv + rr * 128 + cc) =
            *(const v8h*)(kh + (size_t)(b * SK + rr) * ld + hcol + cc);
    }
    __syncthreads();

    const int qrow0 = qt * 128 + wave * 16;                // within this batch
    const size_t qbase = ((size_t)b * SQ + qrow0) * ld + hcol;

    v8f acc[8];
#pragma unroll
    for (int t = 0; t < 8; ++t) acc[t] = v8f{};

#pragma unroll
    for (int ks = 0; ks < 4; ++ks) {
        // A fragment straight from global qh (row-major, K-contiguous)
        v16h af;
        {
            int rr = lane & 15;
            int ko = (lane >> 4) << 3;
            const _Float16* p = qh + qbase + (size_t)rr * ld + ks * 32 + ko;
            F16x16 u;
            u.p.lo = *(const v8h*)p;
            u.p.hi = *(const v8h*)(p + 16);
            af = u.v;
        }
#pragma unroll
        for (int t = 0; t < 8; ++t) {
            v16h bf = frag_b(kv + t * 16 * 128 + ks * 32, 128, lane);
            acc[t] = wmma16(af, bf, acc[t]);
        }
    }

    const float scale = 0.08838834764831845f;   // 1/sqrt(128)
    const int radd = (lane >> 4) << 3;
    const int cn   = lane & 15;

    // scale + additive rh bias
#pragma unroll
    for (int e = 0; e < 8; ++e) {
        const size_t rbase = ((size_t)b * SQ + qrow0 + e + radd) * ld + hcol;
#pragma unroll
        for (int t = 0; t < 8; ++t)
            acc[t][e] = acc[t][e] * scale + (float)rh[rbase + t * 16 + cn];
    }

    // softmax over 128 keys: per-lane over 8 tiles, then 16-lane xor reduce
#pragma unroll
    for (int e = 0; e < 8; ++e) {
        float mx = acc[0][e];
#pragma unroll
        for (int t = 1; t < 8; ++t) mx = fmaxf(mx, acc[t][e]);
#pragma unroll
        for (int s = 8; s >= 1; s >>= 1) mx = fmaxf(mx, __shfl_xor(mx, s, 32));
        float sum = 0.f;
#pragma unroll
        for (int t = 0; t < 8; ++t) {
            float ex = __expf(acc[t][e] - mx);
            acc[t][e] = ex;
            sum += ex;
        }
#pragma unroll
        for (int s = 8; s >= 1; s >>= 1) sum += __shfl_xor(sum, s, 32);
        float inv = 1.0f / sum;
#pragma unroll
        for (int t = 0; t < 8; ++t) acc[t][e] *= inv;
    }

    // spill P (16x128) to per-wave LDS in row-major (A-operand source)
    _Float16* pw = &ps[wave][0];
#pragma unroll
    for (int e = 0; e < 8; ++e) {
        int rm = e + radd;
#pragma unroll
        for (int t = 0; t < 8; ++t)
            pw[rm * 128 + t * 16 + cn] = (_Float16)acc[t][e];
    }
    __syncthreads();   // everyone done reading kh from kv

    // restage vh^T: kv[dcol][key]
    for (int it = 0; it < 8; ++it) {
        int id = tid + it * 256;
        int rr = id >> 4, cc = (id & 15) << 3;   // rr = key, cc.. = d
        v8h vv = *(const v8h*)(vh + (size_t)(b * SK + rr) * ld + hcol + cc);
#pragma unroll
        for (int j = 0; j < 8; ++j)
            kv[(cc + j) * 128 + rr] = vv[j];
    }
    __syncthreads();

    // ctx[16 x 128] = P[16 x 128] * vh[128 x 128]
    v8f acc2[8];
#pragma unroll
    for (int t = 0; t < 8; ++t) acc2[t] = v8f{};
#pragma unroll
    for (int ks = 0; ks < 4; ++ks) {
        v16h af = frag_a(pw + ks * 32, 128, lane);
#pragma unroll
        for (int t = 0; t < 8; ++t) {
            v16h bf = frag_b(kv + t * 16 * 128 + ks * 32, 128, lane);
            acc2[t] = wmma16(af, bf, acc2[t]);
        }
    }

#pragma unroll
    for (int t = 0; t < 8; ++t) {
#pragma unroll
        for (int e = 0; e < 8; ++e) {
            int qr = qrow0 + e + radd;
            ctx[((size_t)b * SQ + qr) * ld + hcol + t * 16 + cn] = (_Float16)acc2[t][e];
        }
    }
}

// ---------------------------------------------------------------------------
extern "C" void kernel_launch(void* const* d_in, const int* in_sizes, int n_in,
                              void* d_out, int out_size, void* d_ws, size_t ws_size,
                              hipStream_t stream) {
    (void)in_sizes; (void)n_in; (void)out_size; (void)ws_size;

    const float* q  = (const float*)d_in[0];
    const float* k  = (const float*)d_in[1];
    const float* v  = (const float*)d_in[2];
    const float* r  = (const float*)d_in[3];
    const float* Wq = (const float*)d_in[4];
    const float* Wk = (const float*)d_in[5];
    const float* Wv = (const float*)d_in[6];
    const float* Wo = (const float*)d_in[7];
    float* out = (float*)d_out;

    const size_t MQ = (size_t)NB * SQ;   // 8192
    const size_t MK = (size_t)NB * SK;   // 512
    const size_t H2 = (size_t)HIDDEN * HIDDEN;

    char* w = (char*)d_ws;
    auto carve = [&](size_t elems) {
        _Float16* p = (_Float16*)w;
        w += ((elems * sizeof(_Float16) + 255) / 256) * 256;
        return p;
    };
    _Float16* qf   = carve(MQ * HIDDEN);
    _Float16* rf   = carve(MQ * HIDDEN);
    _Float16* kf   = carve(MK * HIDDEN);
    _Float16* vf   = carve(MK * HIDDEN);
    _Float16* WqT  = carve(H2);
    _Float16* WkT  = carve(H2);
    _Float16* WvT  = carve(H2);
    _Float16* WoT  = carve(H2);
    _Float16* qhp  = carve(MQ * HIDDEN);
    _Float16* rhp  = carve(MQ * HIDDEN);
    _Float16* khp  = carve(MK * HIDDEN);
    _Float16* vhp  = carve(MK * HIDDEN);
    _Float16* ctxp = carve(MQ * HIDDEN);

    // 1) converts
    {
        int nq = (int)(MQ * HIDDEN);
        int nk = (int)(MK * HIDDEN);
        cvt_f32_f16<<<nq / 1024, 256, 0, stream>>>(q, qf, nq);
        cvt_f32_f16<<<nq / 1024, 256, 0, stream>>>(r, rf, nq);
        cvt_f32_f16<<<nk / 1024, 256, 0, stream>>>(k, kf, nk);
        cvt_f32_f16<<<nk / 1024, 256, 0, stream>>>(v, vf, nk);
        dim3 tb(32, 8), tg(HIDDEN / 32, HIDDEN / 32);
        cvt_transpose<<<tg, tb, 0, stream>>>(Wq, WqT, HIDDEN, HIDDEN);
        cvt_transpose<<<tg, tb, 0, stream>>>(Wk, WkT, HIDDEN, HIDDEN);
        cvt_transpose<<<tg, tb, 0, stream>>>(Wv, WvT, HIDDEN, HIDDEN);
        cvt_transpose<<<tg, tb, 0, stream>>>(Wo, WoT, HIDDEN, HIDDEN);
    }

    // 2) projections (f16 outputs); block tile 128x256
    const int gBig = (int)((MQ / 128) * (HIDDEN / 256));   // 64*16 = 1024
    const int gSml = (int)((MK / 128) * (HIDDEN / 256));   // 4*16  = 64
    gemm_f16_t<_Float16><<<gBig, 256, 0, stream>>>(qf, WqT, qhp, (int)MQ, HIDDEN, HIDDEN);
    gemm_f16_t<_Float16><<<gSml, 256, 0, stream>>>(kf, WkT, khp, (int)MK, HIDDEN, HIDDEN);
    gemm_f16_t<_Float16><<<gSml, 256, 0, stream>>>(vf, WvT, vhp, (int)MK, HIDDEN, HIDDEN);
    gemm_f16_t<_Float16><<<gBig, 256, 0, stream>>>(rf, WoT, rhp, (int)MQ, HIDDEN, HIDDEN);

    // 3) fused attention: B*H*(SQ/128) blocks
    attn_kernel<<<NB * HEADS * (SQ / 128), 256, 0, stream>>>(qhp, khp, vhp, rhp, ctxp);

    // 4) output projection (f32 out)
    gemm_f16_t<float><<<gBig, 256, 0, stream>>>(ctxp, WoT, out, (int)MQ, HIDDEN, HIDDEN);
}
